// TwoAreaRNN_27831388078106
// MI455X (gfx1250) — compile-verified, matrix operand order checked by reading, not run on hardware
//
#include <hip/hip_runtime.h>
#include <hip/hip_bf16.h>

// Two-area RNN, persistent-kernel formulation for gfx1250.
// B=128, T=512, H=512, O=3.  alpha1=alpha2=0.1, noise scale 0.02.
//
// grid = 8 workgroups (16 batch rows each), block = 1024 threads = 32 waves.
// Each wave owns one 16-wide N-tile of H; h1/h2 state tiles live in the
// wave's WMMA accumulator registers across all 512 steps. tanh(h) is shared
// via LDS (bf16, padded stride). Weights are pre-converted once to bf16 in
// WMMA B-fragment blocked layout in d_ws and streamed from L2 each step
// (2MB bf16 total, L2 = 192MB), software-pipelined (double-buffered B frags).

#define Hn   512
#define Bn   128
#define Tn   512
#define LSTR 528                       // LDS row stride (elements), 16-elem pad

typedef __bf16 bf16_t;
typedef bf16_t v16bf __attribute__((ext_vector_type(16)));
typedef bf16_t v8bf  __attribute__((ext_vector_type(8)));
typedef float  v8f   __attribute__((ext_vector_type(8)));

__device__ __forceinline__ v16bf cat8(v8bf lo, v8bf hi) {
  return __builtin_shufflevector(lo, hi, 0,1,2,3,4,5,6,7,8,9,10,11,12,13,14,15);
}

// A-fragment (16x32 bf16) per ISA layout:
// lanes 0-15: M=lane, K = {0..7, 16..23}; lanes 16-31: M=lane-16, K = {8..15, 24..31}
// -> per lane: two contiguous 16B runs from row M at kOff and kOff+16.
__device__ __forceinline__ v16bf ldsA(const bf16_t* t, int M, int kOff) {
  const v8bf* p = (const v8bf*)(t + M * LSTR + kOff);
  return cat8(p[0], p[2]);             // +2 v8bf == +16 elements
}

// B-fragment: 16 contiguous bf16 per lane from the pre-blocked weight buffer.
__device__ __forceinline__ v16bf ldB(const bf16_t* w) {
  const v8bf* p = (const v8bf*)w;
  return cat8(p[0], p[1]);
}

__device__ __forceinline__ v8f wmma_bf16(v16bf a, v16bf b, v8f c) {
  return __builtin_amdgcn_wmma_f32_16x16x32_bf16(false, a, false, b,
                                                 (short)0, c, false, false);
}

// One GEMM phase: acc0 += A1(t1s) @ WA, acc1 += A2(t2s) @ WB over K=512,
// with the global B-fragments double-buffered one kb-iteration ahead so the
// L2 load latency overlaps the WMMA chains instead of serializing them.
__device__ __forceinline__ void gemm_phase(const bf16_t* __restrict__ a1buf,
                                           const bf16_t* __restrict__ a2buf,
                                           const bf16_t* __restrict__ wA,
                                           const bf16_t* __restrict__ wB,
                                           int nl, int base,
                                           v8f& acc0, v8f& acc1) {
  v16bf b1c = ldB(wA);
  v16bf b2c = ldB(wB);
  #pragma unroll
  for (int kb = 0; kb < 15; ++kb) {
    v16bf b1n = ldB(wA + (size_t)(kb + 1) * 512);   // prefetch next kb
    v16bf b2n = ldB(wB + (size_t)(kb + 1) * 512);
    v16bf a1 = ldsA(a1buf, nl, kb * 32 + base);
    v16bf a2 = ldsA(a2buf, nl, kb * 32 + base);
    acc0 = wmma_bf16(a1, b1c, acc0);
    acc1 = wmma_bf16(a2, b2c, acc1);
    b1c = b1n;
    b2c = b2n;
  }
  v16bf a1 = ldsA(a1buf, nl, 15 * 32 + base);
  v16bf a2 = ldsA(a2buf, nl, 15 * 32 + base);
  acc0 = wmma_bf16(a1, b1c, acc0);
  acc1 = wmma_bf16(a2, b2c, acc1);
}

// Convert fp32 row-major [H][H] weights into bf16 blocked B-fragment layout:
// block (nt, kb) holds the 32(K)x16(N) tile for N-tile nt, K-block kb.
// Lane l stores column N = nt*16 + (l&15), K = Kb..Kb+7 then Kb+16..Kb+23,
// Kb = kb*32 + (l>=16 ? 8 : 0)  (mirrors the A-operand K striping).
__global__ void prep_weights(const float* __restrict__ w11,
                             const float* __restrict__ w12,
                             const float* __restrict__ w21,
                             const float* __restrict__ w22,
                             bf16_t* __restrict__ ws) {
  const int lane = threadIdx.x;                 // 0..31
  const int bid  = blockIdx.x;                  // 0..2047
  const int m    = bid >> 9;                    // matrix 0..3
  const int rest = bid & 511;                   // nt*16 + kb
  const int nt   = rest >> 4;
  const int kb   = rest & 15;
  const float* W = (m == 0) ? w11 : (m == 1) ? w12 : (m == 2) ? w21 : w22;
  const int N  = nt * 16 + (lane & 15);
  const int Kb = kb * 32 + ((lane >> 4) << 3);

  v8bf lo, hiv;
  #pragma unroll
  for (int j = 0; j < 8; ++j) {
    lo[j]  = (bf16_t)W[(Kb + j) * Hn + N];
    hiv[j] = (bf16_t)W[(Kb + 16 + j) * Hn + N];
  }
  v8bf* dst = (v8bf*)(ws + (size_t)m * Hn * Hn + ((size_t)rest * 32 + lane) * 16);
  dst[0] = lo;
  dst[1] = hiv;
}

__global__ __launch_bounds__(1024)
void rnn_persistent(const float* __restrict__ x,        // [B,T,5]
                    const float* __restrict__ wi_stim,  // [3,H]
                    const float* __restrict__ wi_ctx,   // [2,H]
                    const float* __restrict__ wo,       // [H,3]
                    const float* __restrict__ noise1,   // [T,B,H]
                    const float* __restrict__ noise2,   // [T,B,H]
                    const bf16_t* __restrict__ wsW,     // blocked bf16 weights
                    float* __restrict__ out)            // [B,T,3]
{
  __shared__ __align__(16) bf16_t t1s[16 * LSTR];   // tanh(h1), bf16
  __shared__ __align__(16) bf16_t t2s[16 * LSTR];   // tanh(h2), bf16
  __shared__ float wo_s[Hn * 3];
  __shared__ float xbuf[16 * 5];
  __shared__ float red[48 * 16];

  const int tid  = threadIdx.x;
  const int lane = tid & 31;
  const int w    = tid >> 5;              // wave id == N-tile id (0..31)
  const int base = (lane >> 4) * 8;       // M-row / K half-offset selector
  const int nl   = lane & 15;
  const int nCol = w * 16 + nl;           // column in H
  const int b0   = blockIdx.x * 16;       // batch base

  // Blocked weight tile pointers for this wave's N-tile (+ per-lane offset).
  const size_t MSZ = (size_t)Hn * Hn;
  const bf16_t* w11t = wsW + 0 * MSZ + (size_t)w * 8192 + (size_t)lane * 16;
  const bf16_t* w12t = wsW + 1 * MSZ + (size_t)w * 8192 + (size_t)lane * 16;
  const bf16_t* w21t = wsW + 2 * MSZ + (size_t)w * 8192 + (size_t)lane * 16;
  const bf16_t* w22t = wsW + 3 * MSZ + (size_t)w * 8192 + (size_t)lane * 16;

  // Init: tanh(0)=0 activations, stage wo into LDS.
  for (int i = tid; i < 16 * LSTR; i += 1024) {
    t1s[i] = (bf16_t)0.f;
    t2s[i] = (bf16_t)0.f;
  }
  for (int i = tid; i < Hn * 3; i += 1024) wo_s[i] = wo[i];

  // Per-lane constant input weights for this column.
  const float ws0 = wi_stim[0 * Hn + nCol];
  const float ws1 = wi_stim[1 * Hn + nCol];
  const float ws2 = wi_stim[2 * Hn + nCol];
  const float wc0 = wi_ctx[0 * Hn + nCol];
  const float wc1 = wi_ctx[1 * Hn + nCol];

  // Recurrent state tiles (C/D layout: VGPR j -> row base+j, col nCol).
  float h1[8] = {0.f, 0.f, 0.f, 0.f, 0.f, 0.f, 0.f, 0.f};
  float h2[8] = {0.f, 0.f, 0.f, 0.f, 0.f, 0.f, 0.f, 0.f};

  __syncthreads();

  #pragma unroll 1
  for (int t = 0; t < Tn; ++t) {
    // Opaque zero offset: defeats LICM so the 64 weight B-fragments are
    // re-streamed from L2 each step instead of being hoisted out of the
    // t-loop and spilled to scratch.
    unsigned woff = 0;
    asm volatile("" : "+s"(woff));
    const bf16_t* w11p = w11t + woff;
    const bf16_t* w12p = w12t + woff;
    const bf16_t* w21p = w21t + woff;
    const bf16_t* w22p = w22t + woff;

    // Stage this step's stimulus/context rows.
    if (tid < 80)
      xbuf[tid] = x[(size_t)(b0 + tid / 5) * Tn * 5 + (size_t)t * 5 + (tid % 5)];

    // Stream phase-1 noise, non-temporal (read exactly once overall).
    float n1v[8];
    #pragma unroll
    for (int j = 0; j < 8; ++j)
      n1v[j] = __builtin_nontemporal_load(
          &noise1[((size_t)t * Bn + b0 + base + j) * Hn + nCol]);

    __syncthreads();                                   // xbuf + prior t1/t2 visible

    // ---- Phase 1: acc = tanh(h1)@W11 + tanh(h2)@W12 ----
    v8f accA = {}, accB = {};
    gemm_phase(t1s, t2s, w11p, w12p, nl, base, accA, accB);

    #pragma unroll
    for (int j = 0; j < 8; ++j) {
      const int Mj = base + j;
      const float stim = xbuf[Mj * 5 + 0] * ws0 + xbuf[Mj * 5 + 1] * ws1 +
                         xbuf[Mj * 5 + 2] * ws2;
      const float r = accA[j] + accB[j] + stim + 0.02f * n1v[j];
      h1[j] = 0.9f * h1[j] + 0.1f * r;
    }
    __syncthreads();                                   // all phase-1 LDS reads done

    #pragma unroll
    for (int j = 0; j < 8; ++j)
      t1s[(base + j) * LSTR + nCol] = (bf16_t)tanhf(h1[j]);

    float n2v[8];
    #pragma unroll
    for (int j = 0; j < 8; ++j)
      n2v[j] = __builtin_nontemporal_load(
          &noise2[((size_t)t * Bn + b0 + base + j) * Hn + nCol]);

    __syncthreads();                                   // new tanh(h1) visible

    // ---- Phase 2: acc = tanh(h1_new)@W21 + tanh(h2_old)@W22 ----
    v8f accC = {}, accD = {};
    gemm_phase(t1s, t2s, w21p, w22p, nl, base, accC, accD);

    #pragma unroll
    for (int j = 0; j < 8; ++j) {
      const int Mj = base + j;
      const float ctx = xbuf[Mj * 5 + 3] * wc0 + xbuf[Mj * 5 + 4] * wc1;
      const float r = accC[j] + accD[j] + ctx + 0.02f * n2v[j];
      h2[j] = 0.9f * h2[j] + 0.1f * r;
    }
    __syncthreads();                                   // all phase-2 reads of t2 done

    #pragma unroll
    for (int j = 0; j < 8; ++j)
      t2s[(base + j) * LSTR + nCol] = (bf16_t)tanhf(h2[j]);

    __syncthreads();                                   // new tanh(h2) visible

    // ---- Output: out[b,t,:] = tanh(h2) @ wo  (16x512 @ 512x3) ----
    {
      const int g = tid >> 4, s = tid & 15;
      if (g < 48) {
        const int r = g / 3, o = g - 3 * (g / 3);
        float p = 0.f;
        #pragma unroll 8
        for (int i = 0; i < 32; ++i)
          p += (float)t2s[r * LSTR + s * 32 + i] * wo_s[(s * 32 + i) * 3 + o];
        red[g * 16 + s] = p;
      }
      __syncthreads();
      if (tid < 48) {
        const int r = tid / 3, o = tid - 3 * (tid / 3);
        float sum = 0.f;
        #pragma unroll
        for (int i = 0; i < 16; ++i) sum += red[tid * 16 + i];
        __builtin_nontemporal_store(sum,
            &out[((size_t)(b0 + r) * Tn + t) * 3 + o]);
      }
    }
    __syncthreads();                                   // safe reuse next iteration
  }
}

extern "C" void kernel_launch(void* const* d_in, const int* in_sizes, int n_in,
                              void* d_out, int out_size, void* d_ws, size_t ws_size,
                              hipStream_t stream) {
  (void)in_sizes; (void)n_in; (void)out_size; (void)ws_size;
  const float* x       = (const float*)d_in[0];
  const float* wi_stim = (const float*)d_in[1];
  const float* wrec11  = (const float*)d_in[2];
  const float* wi_ctx  = (const float*)d_in[3];
  const float* wrec22  = (const float*)d_in[4];
  const float* wrec12  = (const float*)d_in[5];
  const float* wrec21  = (const float*)d_in[6];
  const float* wo      = (const float*)d_in[7];
  const float* noise1  = (const float*)d_in[8];
  const float* noise2  = (const float*)d_in[9];
  bf16_t* wsW = (bf16_t*)d_ws;          // 4 * 512*512 bf16 = 2 MB
  float*  out = (float*)d_out;

  // One-time (per launch, deterministic) fp32 -> bf16 blocked weight convert.
  prep_weights<<<2048, 32, 0, stream>>>(wrec11, wrec12, wrec21, wrec22, wsW);

  // Persistent RNN: 8 workgroups x 1024 threads (32 waves), full T loop inside.
  rnn_persistent<<<8, 1024, 0, stream>>>(x, wi_stim, wi_ctx, wo,
                                         noise1, noise2, wsW, out);
}
